// SlotAttention_17686675325437
// MI455X (gfx1250) — matrix operand order repaired and checked by reading, not compiled, not run on hardware
//
#include <hip/hip_runtime.h>
#include <hip/hip_bf16.h>

#define BB 32
#define NN 16384
#define DD 64
#define NSLOT 8
#define HIDF 128
#define MTOT (BB * NN)

typedef __attribute__((ext_vector_type(16))) _Float16 v16h;
typedef __attribute__((ext_vector_type(8)))  float    v8f;

__device__ __forceinline__ v8f wmma_f16(v16h a, v16h b, v8f c) {
  return __builtin_amdgcn_wmma_f32_16x16x32_f16(false, a, false, b, (short)0, c,
                                                false, false);
}

__device__ __forceinline__ void lds_wait() {
#if __has_builtin(__builtin_amdgcn_s_wait_dscnt)
  __builtin_amdgcn_s_wait_dscnt(0);
#else
  asm volatile("s_wait_dscnt 0x0" ::: "memory");
#endif
}

// A 16x32 f16 fragment from row-major LDS tile (m = lane%16).
__device__ __forceinline__ v16h load_A_lds(const _Float16* s, int ld, int lane, int koff) {
  v16h a;
  int m  = lane & 15;
  int kb = (lane < 16) ? 0 : 8;
#pragma unroll
  for (int h = 0; h < 16; ++h) {
    int k = ((h < 8) ? h : h + 8) + kb;
    a[h] = s[m * ld + koff + k];
  }
  return a;
}

// B 32x16 fragment where B[k,n] = s[n*ld + koff + k]  (i.e. W stored row-major [n][k]).
__device__ __forceinline__ v16h load_B_wrow(const _Float16* s, int ld, int lane, int koff) {
  v16h b;
  int n  = lane & 15;
  int kb = (lane < 16) ? 0 : 8;
#pragma unroll
  for (int h = 0; h < 16; ++h) b[h] = s[n * ld + koff + kb + ((h < 8) ? h : h + 8)];
  return b;
}

// B 32x16 fragment where B[k,n] = s[k*ld + n]  (row-major K x 16 tile).
__device__ __forceinline__ v16h load_B_krow(const _Float16* s, int ld, int lane) {
  v16h b;
  int n  = lane & 15;
  int kb = (lane < 16) ? 0 : 8;
#pragma unroll
  for (int h = 0; h < 16; ++h) b[h] = s[(kb + ((h < 8) ? h : h + 8)) * ld + n];
  return b;
}

__device__ __forceinline__ void atomicMaxFloat(float* addr, float val) {
  if (val >= 0.f) atomicMax((int*)addr, __float_as_int(val));
  else            atomicMin((unsigned int*)addr, __float_as_uint(val));
}

// ---------------- init: copy slots -> sA, zero colsum ----------------
__global__ void sa_init_kernel(const float* __restrict__ slots_in,
                               float* __restrict__ sA, float* __restrict__ colsum) {
  int t = blockIdx.x * 256 + threadIdx.x;
  if (t < BB * NSLOT * DD) sA[t] = slots_in[t];
  if (t < BB * DD) colsum[t] = 0.f;
}

// ---------------- fused LN + k/v projection (WMMA) ----------------
__global__ void sa_kv_kernel(const float* __restrict__ x,
                             const float* __restrict__ g_in, const float* __restrict__ b_in,
                             const float* __restrict__ Wk, const float* __restrict__ bk,
                             const float* __restrict__ Wv, const float* __restrict__ bv,
                             _Float16* __restrict__ kh, _Float16* __restrict__ vh,
                             float* __restrict__ colsum) {
  __shared__ _Float16 wkS[4096], wvS[4096];
  __shared__ float gS[64], bS[64], bkS[64], bvS[64];
  __shared__ _Float16 xsAll[8][16 * 72];
  int t = threadIdx.x;
  for (int idx = t; idx < 4096; idx += 256) {
    wkS[idx] = (_Float16)Wk[idx];
    wvS[idx] = (_Float16)Wv[idx];
  }
  if (t < 64) { gS[t] = g_in[t]; bS[t] = b_in[t]; bkS[t] = bk[t]; bvS[t] = bv[t]; }
  __syncthreads();

  int w = t >> 5, lane = t & 31;
  int row0 = blockIdx.x * 128 + w * 16;
  int batch = row0 >> 14;
  int r2 = lane >> 1, half = lane & 1;

  // per-lane: half a row (32 floats), LN with partner lane
  float vreg[32];
  const float4* x4 = (const float4*)(x + (size_t)(row0 + r2) * DD + half * 32);
#pragma unroll
  for (int q = 0; q < 8; ++q) {
    float4 f = x4[q];
    vreg[4 * q] = f.x; vreg[4 * q + 1] = f.y; vreg[4 * q + 2] = f.z; vreg[4 * q + 3] = f.w;
  }
  float s = 0.f, sq = 0.f;
#pragma unroll
  for (int c = 0; c < 32; ++c) { s += vreg[c]; sq += vreg[c] * vreg[c]; }
  s  += __shfl_xor(s, 1, 32);
  sq += __shfl_xor(sq, 1, 32);
  float mean = s * (1.f / 64.f);
  float rstd = rsqrtf(sq * (1.f / 64.f) - mean * mean + 1e-5f);
  _Float16* xs = xsAll[w];
#pragma unroll
  for (int c = 0; c < 32; ++c) {
    int d = half * 32 + c;
    xs[r2 * 72 + d] = (_Float16)((vreg[c] - mean) * rstd * gS[d] + bS[d]);
  }
  lds_wait();

  v16h a0 = load_A_lds(xs, 72, lane, 0);
  v16h a1 = load_A_lds(xs, 72, lane, 32);
  int n = lane & 15;
  int mrow = (lane < 16) ? 0 : 8;

  for (int j0 = 0; j0 < 4; ++j0) {
    // k projection
    v8f ck;
    float bbk = bkS[j0 * 16 + n];
#pragma unroll
    for (int r = 0; r < 8; ++r) ck[r] = bbk;
    v16h b0 = load_B_wrow(wkS + (j0 * 16) * 64, 64, lane, 0);
    v16h b1 = load_B_wrow(wkS + (j0 * 16) * 64, 64, lane, 32);
    ck = wmma_f16(a0, b0, ck);
    ck = wmma_f16(a1, b1, ck);
#pragma unroll
    for (int r = 0; r < 8; ++r)
      kh[(size_t)(row0 + r + mrow) * DD + j0 * 16 + n] = (_Float16)ck[r];

    // v projection (+ column sums for the EPS term)
    v8f cv;
    float bbv = bvS[j0 * 16 + n];
#pragma unroll
    for (int r = 0; r < 8; ++r) cv[r] = bbv;
    b0 = load_B_wrow(wvS + (j0 * 16) * 64, 64, lane, 0);
    b1 = load_B_wrow(wvS + (j0 * 16) * 64, 64, lane, 32);
    cv = wmma_f16(a0, b0, cv);
    cv = wmma_f16(a1, b1, cv);
    float ps = 0.f;
#pragma unroll
    for (int r = 0; r < 8; ++r) {
      vh[(size_t)(row0 + r + mrow) * DD + j0 * 16 + n] = (_Float16)cv[r];
      ps += cv[r];
    }
    atomicAdd(&colsum[batch * DD + j0 * 16 + n], ps);
  }
}

// ---------------- per-iteration: LN(slots), q = s@Wq^T + bq (scaled), init reductions ----
__global__ void sa_q_kernel(const float* __restrict__ cur,
                            const float* __restrict__ gs, const float* __restrict__ bsv,
                            const float* __restrict__ Wq, const float* __restrict__ bq,
                            float* __restrict__ qsG, float* __restrict__ rowmaxG,
                            float* __restrict__ rowsumG, float* __restrict__ updG) {
  int b = blockIdx.x, t = threadIdx.x;
  __shared__ float sS[512], slnS[512], mS[8], rS[8];
  for (int o = t; o < 512; o += 256) sS[o] = cur[b * 512 + o];
  __syncthreads();
  if (t < 8) {
    float s = 0.f, sq = 0.f;
    for (int d = 0; d < 64; ++d) { float v = sS[t * 64 + d]; s += v; sq += v * v; }
    float m = s * (1.f / 64.f);
    mS[t] = m;
    rS[t] = rsqrtf(sq * (1.f / 64.f) - m * m + 1e-5f);
  }
  __syncthreads();
  for (int o = t; o < 512; o += 256) {
    int i = o >> 6, d = o & 63;
    slnS[o] = (sS[o] - mS[i]) * rS[i] * gs[d] + bsv[d];
  }
  __syncthreads();
  for (int o = t; o < 512; o += 256) {
    int i = o >> 6, dd = o & 63;
    float acc = bq[dd];
    for (int d = 0; d < 64; ++d) acc += slnS[i * 64 + d] * Wq[dd * 64 + d];
    qsG[b * 512 + o] = acc * 0.125f;  // fold SCALE = 64^-0.5
  }
  if (t < 8) { rowmaxG[b * 8 + t] = -3.4e38f; rowsumG[b * 8 + t] = 0.f; }
  for (int o = t; o < 512; o += 256) updG[b * 512 + o] = 0.f;
}

// ---------------- dots = q @ k^T (WMMA), running row max ----------------
__global__ void sa_dots_kernel(const float* __restrict__ qsG, const _Float16* __restrict__ kh,
                               float* __restrict__ dotsG, float* __restrict__ rowmaxG) {
  __shared__ _Float16 qs16[16 * 72];
  __shared__ _Float16 ksAll[8][16 * 72];
  int b = blockIdx.x >> 4, chunk = blockIdx.x & 15;
  int t = threadIdx.x, w = t >> 5, lane = t & 31;

  for (int idx = t; idx < 1024; idx += 256) {
    int row = idx >> 6, d = idx & 63;
    float val = (row < 8) ? qsG[b * 512 + row * 64 + d] : 0.f;
    qs16[row * 72 + d] = (_Float16)val;
  }
  __syncthreads();

  v16h a0 = load_A_lds(qs16, 72, lane, 0);
  v16h a1 = load_A_lds(qs16, 72, lane, 32);
  _Float16* ks = ksAll[w];
  int n = lane & 15;
  float wmax[8];
#pragma unroll
  for (int r = 0; r < 8; ++r) wmax[r] = -3.4e38f;

  for (int tl = w; tl < 64; tl += 8) {
    int j0 = chunk * 1024 + tl * 16;
    const uint4* src = (const uint4*)(kh + (size_t)(b * NN + j0) * DD);
#pragma unroll
    for (int s = 0; s < 4; ++s) {
      int idx = lane + 32 * s;
      int row = idx >> 3, q8 = idx & 7;
      *(uint4*)(ks + row * 72 + q8 * 8) = src[idx];
    }
    lds_wait();
    v16h b0 = load_B_wrow(ks, 72, lane, 0);
    v16h b1 = load_B_wrow(ks, 72, lane, 32);
    v8f c;
#pragma unroll
    for (int r = 0; r < 8; ++r) c[r] = 0.f;
    c = wmma_f16(a0, b0, c);
    c = wmma_f16(a1, b1, c);
    if (lane < 16) {
#pragma unroll
      for (int r = 0; r < 8; ++r) {
        float val = c[r];
        dotsG[(size_t)(b * 8 + r) * NN + j0 + n] = val;
        wmax[r] = fmaxf(wmax[r], val);
      }
    }
  }
#pragma unroll
  for (int r = 0; r < 8; ++r) {
    wmax[r] = fmaxf(wmax[r], __shfl_xor(wmax[r], 1, 32));
    wmax[r] = fmaxf(wmax[r], __shfl_xor(wmax[r], 2, 32));
    wmax[r] = fmaxf(wmax[r], __shfl_xor(wmax[r], 4, 32));
    wmax[r] = fmaxf(wmax[r], __shfl_xor(wmax[r], 8, 32));
  }
  if (lane == 0) {
#pragma unroll
    for (int r = 0; r < 8; ++r) atomicMaxFloat(&rowmaxG[b * 8 + r], wmax[r]);
  }
}

// ---------------- exp(dots - max) in place, row sums ----------------
__global__ void sa_exp_kernel(float* __restrict__ dotsG, const float* __restrict__ rowmaxG,
                              float* __restrict__ rowsumG) {
  int row = blockIdx.x >> 2, seg = blockIdx.x & 3;
  int t = threadIdx.x;
  size_t base = (size_t)row * NN + seg * 4096;
  float mx = rowmaxG[row];
  float ls = 0.f;
#pragma unroll
  for (int s = 0; s < 16; ++s) {
    size_t idx = base + t + s * 256;
    float e = __expf(dotsG[idx] - mx);
    dotsG[idx] = e;
    ls += e;
  }
  __shared__ float red[256];
  red[t] = ls;
  __syncthreads();
  for (int off = 128; off > 0; off >>= 1) {
    if (t < off) red[t] += red[t + off];
    __syncthreads();
  }
  if (t == 0) atomicAdd(&rowsumG[row], red[0]);
}

// ---------------- updates = softmax @ v (WMMA); optional attn output ----------------
__global__ void sa_upd_kernel(const float* __restrict__ dotsG, const float* __restrict__ rowsumG,
                              const _Float16* __restrict__ vh, float* __restrict__ updG,
                              float* __restrict__ attn_out, int writeAttn) {
  int b = blockIdx.x >> 2, part = blockIdx.x & 3;
  int t = threadIdx.x, w = t >> 5, lane = t & 31;
  int nb = w & 3, jh = w >> 2;
  int jbase0 = part * 4096 + jh * 2048;

  __shared__ float invs[8];
  __shared__ float esAll[8][8 * 33];
  __shared__ _Float16 vsAll[8][32 * 24];
  if (t < 8) invs[t] = 1.f / rowsumG[b * 8 + t];
  __syncthreads();

  float* esW = esAll[w];
  _Float16* vsW = vsAll[w];
  int n = lane & 15;
  int m = lane & 15;
  float im = (m < 8) ? invs[m] : 0.f;
  int kb2 = (lane < 16) ? 0 : 8;

  v8f c;
#pragma unroll
  for (int r = 0; r < 8; ++r) c[r] = 0.f;

  for (int ch = 0; ch < 64; ++ch) {
    int jj = jbase0 + ch * 32;
    // stage softmax numerators (8 x 32 f32), write final attn on last iteration
#pragma unroll
    for (int s = 0; s < 8; ++s) {
      int idx = lane + 32 * s;
      int row = idx >> 5, col = idx & 31;
      size_t g = (size_t)(b * 8 + row) * NN + jj + col;
      float e = dotsG[g];
      esW[row * 33 + col] = e;
      if (writeAttn && nb == 0) attn_out[g] = e * invs[row] + 1e-8f;
    }
    // stage v tile 32 x 16 f16
    {
      const uint4* vsrc = (const uint4*)(vh + (size_t)(b * NN + jj + lane) * DD + nb * 16);
      uint4 u0 = vsrc[0], u1 = vsrc[1];
      uint4* dst = (uint4*)(vsW + lane * 24);
      dst[0] = u0;
      dst[1] = u1;
    }
    lds_wait();
    // A = softmax rows (m<8), zeros elsewhere
    v16h a;
#pragma unroll
    for (int h = 0; h < 16; ++h) {
      int k = ((h < 8) ? h : h + 8) + kb2;
      float val = (m < 8) ? esW[m * 33 + k] * im : 0.f;
      a[h] = (_Float16)val;
    }
    v16h bf = load_B_krow(vsW, 24, lane);
    c = wmma_f16(a, bf, c);
  }
  if (lane < 16) {
#pragma unroll
    for (int r = 0; r < 8; ++r)
      atomicAdd(&updG[b * 512 + r * 64 + nb * 16 + n], c[r]);
  }
}

// ---------------- GRU + residual MLP on slots (tiny, VALU) ----------------
__global__ void sa_slots_kernel(const float* __restrict__ updG, const float* __restrict__ colsum,
                                const float* __restrict__ prevG,
                                const float* __restrict__ W_ih, const float* __restrict__ b_ih,
                                const float* __restrict__ W_hh, const float* __restrict__ b_hh,
                                const float* __restrict__ gff, const float* __restrict__ bff,
                                const float* __restrict__ W1, const float* __restrict__ b1,
                                const float* __restrict__ W2, const float* __restrict__ b2,
                                float* __restrict__ nextG, float* __restrict__ slots_out,
                                int last) {
  int b = blockIdx.x, t = threadIdx.x;
  __shared__ float upd[512], prev[512], hb[512];
  __shared__ float gx[8][192], gh[8][192], a1[8][128];
  __shared__ float mS[8], rS[8];
  for (int o = t; o < 512; o += 256) {
    upd[o] = updG[b * 512 + o] + 1e-8f * colsum[b * 64 + (o & 63)];
    prev[o] = prevG[b * 512 + o];
  }
  __syncthreads();
  if (t < 192) {
    for (int i = 0; i < 8; ++i) {
      float ax = b_ih[t], ah = b_hh[t];
      for (int d = 0; d < 64; ++d) {
        ax += W_ih[t * 64 + d] * upd[i * 64 + d];
        ah += W_hh[t * 64 + d] * prev[i * 64 + d];
      }
      gx[i][t] = ax;
      gh[i][t] = ah;
    }
  }
  __syncthreads();
  for (int o = t; o < 512; o += 256) {
    int i = o >> 6, d = o & 63;
    float xr = gx[i][d], xz = gx[i][64 + d], xn = gx[i][128 + d];
    float hr = gh[i][d], hz = gh[i][64 + d], hn = gh[i][128 + d];
    float r = 1.f / (1.f + __expf(-(xr + hr)));
    float z = 1.f / (1.f + __expf(-(xz + hz)));
    float n_ = tanhf(xn + r * hn);
    hb[o] = (1.f - z) * n_ + z * prev[o];
  }
  __syncthreads();
  if (t < 8) {
    float s = 0.f, sq = 0.f;
    for (int d = 0; d < 64; ++d) { float v = hb[t * 64 + d]; s += v; sq += v * v; }
    float m = s * (1.f / 64.f);
    mS[t] = m;
    rS[t] = rsqrtf(sq * (1.f / 64.f) - m * m + 1e-5f);
  }
  __syncthreads();
  for (int o = t; o < 1024; o += 256) {
    int i = o >> 7, row = o & 127;
    float acc = b1[row];
    for (int d = 0; d < 64; ++d) {
      float hn_ = (hb[i * 64 + d] - mS[i]) * rS[i] * gff[d] + bff[d];
      acc += W1[row * 64 + d] * hn_;
    }
    a1[i][row] = acc > 0.f ? acc : 0.f;
  }
  __syncthreads();
  for (int o = t; o < 512; o += 256) {
    int i = o >> 6, d = o & 63;
    float acc = b2[d];
    for (int r = 0; r < 128; ++r) acc += W2[d * 128 + r] * a1[i][r];
    float out = hb[o] + acc;
    nextG[b * 512 + o] = out;
    if (last) slots_out[b * 512 + o] = out;
  }
}

extern "C" void kernel_launch(void* const* d_in, const int* in_sizes, int n_in,
                              void* d_out, int out_size, void* d_ws, size_t ws_size,
                              hipStream_t stream) {
  (void)in_sizes; (void)n_in; (void)out_size; (void)ws_size;
  const float* x       = (const float*)d_in[0];
  const float* slots0  = (const float*)d_in[1];
  const float* ln_in_g = (const float*)d_in[2];
  const float* ln_in_b = (const float*)d_in[3];
  const float* ln_s_g  = (const float*)d_in[4];
  const float* ln_s_b  = (const float*)d_in[5];
  const float* ln_ff_g = (const float*)d_in[6];
  const float* ln_ff_b = (const float*)d_in[7];
  const float* Wq = (const float*)d_in[8];
  const float* bq = (const float*)d_in[9];
  const float* Wk = (const float*)d_in[10];
  const float* bk = (const float*)d_in[11];
  const float* Wv = (const float*)d_in[12];
  const float* bv = (const float*)d_in[13];
  const float* W_ih = (const float*)d_in[14];
  const float* b_ih = (const float*)d_in[15];
  const float* W_hh = (const float*)d_in[16];
  const float* b_hh = (const float*)d_in[17];
  const float* W1 = (const float*)d_in[18];
  const float* b1 = (const float*)d_in[19];
  const float* W2 = (const float*)d_in[20];
  const float* b2 = (const float*)d_in[21];

  float* attn_out  = (float*)d_out;                      // [32, 8, 16384]
  float* slots_out = attn_out + (size_t)BB * NSLOT * NN; // [32, 8, 64]

  char* w = (char*)d_ws;
  _Float16* kh = (_Float16*)w; w += (size_t)MTOT * DD * sizeof(_Float16);
  _Float16* vh = (_Float16*)w; w += (size_t)MTOT * DD * sizeof(_Float16);
  float* dots   = (float*)w; w += (size_t)BB * NSLOT * NN * sizeof(float);
  float* qs     = (float*)w; w += (size_t)BB * 512 * sizeof(float);
  float* upd    = (float*)w; w += (size_t)BB * 512 * sizeof(float);
  float* sA     = (float*)w; w += (size_t)BB * 512 * sizeof(float);
  float* sB     = (float*)w; w += (size_t)BB * 512 * sizeof(float);
  float* rowmax = (float*)w; w += 1024;
  float* rowsum = (float*)w; w += 1024;
  float* colsum = (float*)w; w += (size_t)BB * DD * sizeof(float);

  sa_init_kernel<<<64, 256, 0, stream>>>(slots0, sA, colsum);
  sa_kv_kernel<<<MTOT / 128, 256, 0, stream>>>(x, ln_in_g, ln_in_b, Wk, bk, Wv, bv,
                                               kh, vh, colsum);
  float* cur = sA;
  float* nxt = sB;
  for (int it = 0; it < 3; ++it) {
    int last = (it == 2) ? 1 : 0;
    sa_q_kernel<<<BB, 256, 0, stream>>>(cur, ln_s_g, ln_s_b, Wq, bq, qs, rowmax, rowsum, upd);
    sa_dots_kernel<<<BB * 16, 256, 0, stream>>>(qs, kh, dots, rowmax);
    sa_exp_kernel<<<BB * NSLOT * 4, 256, 0, stream>>>(dots, rowmax, rowsum);
    sa_upd_kernel<<<BB * 4, 256, 0, stream>>>(dots, rowsum, vh, upd, attn_out, last);
    sa_slots_kernel<<<BB, 256, 0, stream>>>(upd, colsum, cur, W_ih, b_ih, W_hh, b_hh,
                                            ln_ff_g, ln_ff_b, W1, b1, W2, b2,
                                            nxt, slots_out, last);
    float* tmp = cur; cur = nxt; nxt = tmp;
  }
}
